// FCMNet_39779987096393
// MI455X (gfx1250) — matrix operand order repaired and checked by reading, not compiled, not run on hardware
//
#include <hip/hip_runtime.h>

typedef __attribute__((ext_vector_type(16))) _Float16 v16h;
typedef __attribute__((ext_vector_type(8)))  float    v8f;

// ---------------------------------------------------------------------------
// WMMA GEMM:  C[M,N] = A[M,K] @ B[K,N]  (row-major, packed; f32 io, f16 MACs,
// f32 accumulate). One wave per (16*TM) x (16*TN) output tile; A fragments
// reused across TN column tiles, B fragments across TM row tiles.
// Batched via blockIdx.z (sA may be 0 to share A across batch).
// Row/col out-of-range handled by pointer clamping (garbage lanes never
// stored); K handled by full-tile main loop (constant-offset loads from
// stepped base pointers) + one branchless masked tail.
// ---------------------------------------------------------------------------
template<int TM, int TN>
__global__ __launch_bounds__(32)
void gemm_wmma(const float* __restrict__ A,
               const float* __restrict__ B,
               float* __restrict__ C,
               int M, int N, int K,
               long long sA, long long sB, long long sC)
{
    const float* Ab = A + (size_t)blockIdx.z * (size_t)sA;
    const float* Bb = B + (size_t)blockIdx.z * (size_t)sB;
    float*       Cb = C + (size_t)blockIdx.z * (size_t)sC;

    const int row0 = blockIdx.y * (TM * 16);
    const int col0 = blockIdx.x * (TN * 16);
    const int lane = threadIdx.x & 31;
    const int mn   = lane & 15;
    const bool hi  = lane >= 16;
    const int kba  = hi ? 16 : 0;   // A frag: lanes<16 K{0..15}; hi lanes K{16..31}
    const int koff = hi ? 16 : 0;   // B frag: halves j -> K = koff+j

    int arow[TM];
    const float* Ar0[TM];   // row base (k=0)
    const float* Aq[TM];    // stepped: row base + kba
#pragma unroll
    for (int m = 0; m < TM; ++m) {
        arow[m] = row0 + m * 16 + mn;
        int rc  = arow[m] < M ? arow[m] : (M - 1);      // clamp; discarded at store
        Ar0[m]  = Ab + (size_t)rc * (size_t)K;
        Aq[m]   = Ar0[m] + kba;
    }

    int bcol[TN];
    const float* Bp0[TN];   // column base (k=0)
    const float* Bq[TN];    // stepped: column base + koff*N
#pragma unroll
    for (int t = 0; t < TN; ++t) {
        bcol[t] = col0 + t * 16 + mn;
        int bc  = bcol[t] < N ? bcol[t] : (N - 1);      // clamp; discarded at store
        Bp0[t]  = Bb + (size_t)bc;
        Bq[t]   = Bp0[t] + (size_t)koff * (size_t)N;
    }

    v8f acc[TM][TN];
#pragma unroll
    for (int m = 0; m < TM; ++m)
#pragma unroll
        for (int t = 0; t < TN; ++t) acc[m][t] = (v8f){};

    const int    kFull = K & ~31;
    const size_t bstep = 32u * (size_t)N;

    for (int k0 = 0; k0 < kFull; k0 += 32) {
        v16h af[TM];
#pragma unroll
        for (int m = 0; m < TM; ++m) {
#pragma unroll
            for (int j = 0; j < 8; ++j) {
                af[m][j]     = (_Float16)Aq[m][j];        // constant instr offsets
                af[m][j + 8] = (_Float16)Aq[m][8 + j];
            }
            __builtin_prefetch(Aq[m] + 64, 0, 0);         // next A tiles -> global_prefetch
            Aq[m] += 32;
        }
#pragma unroll
        for (int t = 0; t < TN; ++t) {
            v16h bf;
#pragma unroll
            for (int j = 0; j < 16; ++j)
                bf[j] = (_Float16)Bq[t][(size_t)j * (size_t)N];  // constant instr offsets
            Bq[t] += bstep;
#pragma unroll
            for (int m = 0; m < TM; ++m)
                acc[m][t] = __builtin_amdgcn_wmma_f32_16x16x32_f16(false, af[m], false, bf,
                                                                   (short)0, acc[m][t], false, false);
        }
    }

    if (kFull < K) {   // single branchless masked tail tile
        v16h af[TM];
#pragma unroll
        for (int m = 0; m < TM; ++m)
#pragma unroll
            for (int j = 0; j < 8; ++j) {
                int k1 = kFull + kba + j, k2 = k1 + 8;
                float a1 = Ar0[m][k1 < K ? k1 : 0];
                float a2 = Ar0[m][k2 < K ? k2 : 0];
                af[m][j]     = (_Float16)(k1 < K ? a1 : 0.0f);
                af[m][j + 8] = (_Float16)(k2 < K ? a2 : 0.0f);
            }
#pragma unroll
        for (int t = 0; t < TN; ++t) {
            v16h bf;
#pragma unroll
            for (int j = 0; j < 16; ++j) {
                int k = kFull + koff + j;
                float bv = Bp0[t][(size_t)(k < K ? k : 0) * (size_t)N];
                bf[j] = (_Float16)(k < K ? bv : 0.0f);
            }
#pragma unroll
            for (int m = 0; m < TM; ++m)
                acc[m][t] = __builtin_amdgcn_wmma_f32_16x16x32_f16(false, af[m], false, bf,
                                                                   (short)0, acc[m][t], false, false);
        }
    }

#pragma unroll
    for (int m = 0; m < TM; ++m) {
        const int rbase = row0 + m * 16 + (hi ? 8 : 0);  // C/D layout: VGPR r -> M = rbase + r
#pragma unroll
        for (int t = 0; t < TN; ++t) {
            if (bcol[t] < N) {
                float* cp = Cb + (size_t)rbase * (size_t)N + bcol[t];
#pragma unroll
                for (int r = 0; r < 8; ++r)
                    if (rbase + r < M) cp[(size_t)r * (size_t)N] = acc[m][t][r];
            }
        }
    }
}

// ---------------------------------------------------------------------------
// Edge-conditioned conv message + scatter. One wave per (edge, 16-col tile).
// Batch B=16 == WMMA M dim:  msg[b, cols] = x[b, src, :] @ k_e[:, cols],
// atomically accumulated into out[b, dst, col].
// ---------------------------------------------------------------------------
__global__ __launch_bounds__(32)
void edge_msg_wmma(const float* __restrict__ x,
                   const int* __restrict__ ei,     // [2,E]: src then dst
                   const float* __restrict__ kbuf, // [nE, Ci, Co]
                   float* __restrict__ out,        // [16, Nn, Co]
                   int E, int e0, int nE, int Nn, int Ci, int Co, int tiles)
{
    const int gid = blockIdx.x;
    const int eL  = gid / tiles;
    const int t   = gid - eL * tiles;
    if (eL >= nE) return;
    const int src = ei[e0 + eL];
    const int dst = ei[E + e0 + eL];

    const int lane = threadIdx.x & 31;
    const int mn   = lane & 15;          // batch idx (A rows) and col idx (B/C)
    const bool hi  = lane >= 16;
    const int kba  = hi ? 16 : 0;
    const int koff = hi ? 16 : 0;

    const float* xs = x + ((size_t)mn * (size_t)Nn + (size_t)src) * (size_t)Ci;
    const float* kb = kbuf + (size_t)eL * (size_t)Ci * (size_t)Co;
    const int  bcol = (t << 4) + mn;
    const int  bc   = bcol < Co ? bcol : (Co - 1);
    const float* Bp = kb + bc;

    const float* Aq = xs + kba;
    const float* Bq = Bp + (size_t)koff * (size_t)Co;
    const size_t bstep = 32u * (size_t)Co;

    v8f acc = {};
    const int kFull = Ci & ~31;
    for (int k0 = 0; k0 < kFull; k0 += 32) {
        v16h af, bf;
#pragma unroll
        for (int j = 0; j < 8; ++j) {
            af[j]     = (_Float16)Aq[j];
            af[j + 8] = (_Float16)Aq[8 + j];
        }
#pragma unroll
        for (int j = 0; j < 16; ++j)
            bf[j] = (_Float16)Bq[(size_t)j * (size_t)Co];
        Aq += 32;
        Bq += bstep;
        acc = __builtin_amdgcn_wmma_f32_16x16x32_f16(false, af, false, bf,
                                                     (short)0, acc, false, false);
    }
    if (kFull < Ci) {   // only for Ci=3 level
        v16h af, bf;
#pragma unroll
        for (int j = 0; j < 8; ++j) {
            int k1 = kFull + kba + j, k2 = k1 + 8;
            float a1 = xs[k1 < Ci ? k1 : 0];
            float a2 = xs[k2 < Ci ? k2 : 0];
            af[j]     = (_Float16)(k1 < Ci ? a1 : 0.0f);
            af[j + 8] = (_Float16)(k2 < Ci ? a2 : 0.0f);
        }
#pragma unroll
        for (int j = 0; j < 16; ++j) {
            int k = kFull + koff + j;
            float bv = Bp[(size_t)(k < Ci ? k : 0) * (size_t)Co];
            bf[j] = (_Float16)(k < Ci ? bv : 0.0f);
        }
        acc = __builtin_amdgcn_wmma_f32_16x16x32_f16(false, af, false, bf,
                                                     (short)0, acc, false, false);
    }

    if (bcol < Co) {
        const int bbase = hi ? 8 : 0;
        float* op = out + (size_t)dst * (size_t)Co + (size_t)bcol;
#pragma unroll
        for (int r = 0; r < 8; ++r)
            unsafeAtomicAdd(op + (size_t)(bbase + r) * (size_t)Nn * (size_t)Co, acc[r]);
    }
}

// out[i] = bias[i % C]   (initializes scatter target with broadcast bias)
__global__ void fill_bias_kernel(float* __restrict__ out, const float* __restrict__ bias,
                                 int C, long long total)
{
    long long i = (long long)blockIdx.x * blockDim.x + threadIdx.x;
    if (i < total) out[i] = bias[(int)(i % (long long)C)];
}

__global__ void relu_kernel(float* __restrict__ p, long long n)
{
    long long i = (long long)blockIdx.x * blockDim.x + threadIdx.x;
    if (i < n) p[i] = fmaxf(p[i], 0.0f);
}

// z[16,8] = xflat[16,6016] @ encW[8,6016]^T + encb
__global__ void enc_kernel(const float* __restrict__ xf, const float* __restrict__ W,
                           const float* __restrict__ b, float* __restrict__ z)
{
    int t = threadIdx.x;
    if (t >= 128) return;
    int bb = t >> 3, j = t & 7;
    const float* xr = xf + (size_t)bb * 6016;
    const float* wr = W  + (size_t)j  * 6016;
    float s = b[j];
    for (int k = 0; k < 6016; ++k) s += xr[k] * wr[k];
    z[t] = s;
}

// y[16,6016] = z[16,8] @ decW[6016,8]^T + decb
__global__ void dec_kernel(const float* __restrict__ z, const float* __restrict__ W,
                           const float* __restrict__ b, float* __restrict__ y)
{
    long long i = (long long)blockIdx.x * blockDim.x + threadIdx.x;
    if (i >= 16LL * 6016LL) return;
    int bb = (int)(i / 6016);
    int m  = (int)(i % 6016);
    float s = b[m];
#pragma unroll
    for (int j = 0; j < 8; ++j) s += z[bb * 8 + j] * W[(size_t)m * 8 + j];
    y[i] = s;
}

// ---------------------------------------------------------------------------
extern "C" void kernel_launch(void* const* d_in, const int* in_sizes, int n_in,
                              void* d_out, int out_size, void* d_ws, size_t ws_size,
                              hipStream_t stream)
{
    (void)in_sizes; (void)n_in; (void)out_size;

    static const int NSh[5] = {12000, 3000, 750, 188, 47};
    static const int ESh[5] = {72000, 18000, 4500, 1128, 282};
    static const int FLh[5] = {3, 32, 64, 128, 128};

    const float* x = (const float*)d_in[0];
    const int*   ei[5]; const float* ec[5];
    for (int i = 0; i < 5; ++i) {
        ei[i] = (const int*)d_in[1 + 2 * i];
        ec[i] = (const float*)d_in[2 + 2 * i];
    }
    const float *Wc[4], *bc[4], *Wd[4], *bd[4], *dm[4], *um[4];
    for (int i = 0; i < 4; ++i) {
        int base = 11 + 6 * i;
        Wc[i] = (const float*)d_in[base + 0];
        bc[i] = (const float*)d_in[base + 1];
        Wd[i] = (const float*)d_in[base + 2];
        bd[i] = (const float*)d_in[base + 3];
        dm[i] = (const float*)d_in[base + 4];
        um[i] = (const float*)d_in[base + 5];
    }
    const float* encW = (const float*)d_in[35];
    const float* encb = (const float*)d_in[36];
    const float* decW = (const float*)d_in[37];
    const float* decb = (const float*)d_in[38];

    // workspace carve: two ping-pong activation buffers + z + edge-kernel chunk buf
    const size_t ACT = (size_t)16 * 12000 * 32;   // max activation (floats)
    float* bufA = (float*)d_ws;
    float* bufB = bufA + ACT;
    float* zbuf = bufB + ACT;                     // 256 floats reserved
    float* kbuf = zbuf + 256;
    size_t usedBytes = (2 * ACT + 256) * sizeof(float);
    long long kcap = (ws_size > usedBytes) ? (long long)((ws_size - usedBytes) / 4) : 0;

    auto launch_gemm = [&](const float* A, const float* B, float* C, int M, int N, int K,
                           long long sA, long long sB, long long sC, unsigned batches) {
        if (N >= 64) {
            dim3 g((unsigned)((N + 63) / 64), (unsigned)((M + 31) / 32), batches);
            gemm_wmma<2, 4><<<g, 32, 0, stream>>>(A, B, C, M, N, K, sA, sB, sC);
        } else if (N >= 32) {
            dim3 g((unsigned)((N + 31) / 32), (unsigned)((M + 31) / 32), batches);
            gemm_wmma<2, 2><<<g, 32, 0, stream>>>(A, B, C, M, N, K, sA, sB, sC);
        } else {
            dim3 g((unsigned)((N + 15) / 16), (unsigned)((M + 15) / 16), batches);
            gemm_wmma<1, 1><<<g, 32, 0, stream>>>(A, B, C, M, N, K, sA, sB, sC);
        }
    };

    auto conv = [&](const float* xin, int lvl, const float* W, const float* bias,
                    int Ci, int Co, float* out, bool doRelu) {
        const int Nn = NSh[lvl], E = ESh[lvl];
        const long long total = 16LL * Nn * Co;
        fill_bias_kernel<<<(unsigned)((total + 255) / 256), 256, 0, stream>>>(out, bias, Co, total);

        const long long CiCo = (long long)Ci * Co;
        long long chunk = (kcap > 0) ? (kcap / CiCo) : 1;
        if (chunk < 1) chunk = 1;
        if (chunk > E) chunk = E;
        const int tiles = (Co + 15) >> 4;

        for (int eb = 0; eb < E; eb += (int)chunk) {
            int nE = E - eb; if (nE > chunk) nE = (int)chunk;
            // per-edge kernel mix: kbuf[nE, Ci*Co] = ec[eb:eb+nE, 16] @ Wflat[16, Ci*Co]
            launch_gemm(ec[lvl] + (size_t)eb * 16, W, kbuf, nE, (int)CiCo, 16, 0, 0, 0, 1);
            // message GEMM + scatter
            unsigned nb = (unsigned)(nE * tiles);
            edge_msg_wmma<<<nb, 32, 0, stream>>>(xin, ei[lvl], kbuf, out,
                                                 E, eb, nE, Nn, Ci, Co, tiles);
        }
        if (doRelu)
            relu_kernel<<<(unsigned)((total + 255) / 256), 256, 0, stream>>>(out, total);
    };

    auto pool = [&](const float* P, const float* xin, float* out, int Mo, int Ni, int C) {
        // out[b, Mo, C] = P[Mo, Ni] @ xin[b, Ni, C]   for b = 0..15
        launch_gemm(P, xin, out, Mo, C, Ni, 0, (long long)Ni * C, (long long)Mo * C, 16);
    };

    // ------------------------- encoder -------------------------
    const float* cur = x;
    for (int i = 0; i < 4; ++i) {
        conv(cur, i, Wc[i], bc[i], FLh[i], FLh[i + 1], bufA, true);
        pool(dm[i], bufA, bufB, NSh[i + 1], NSh[i], FLh[i + 1]);
        cur = bufB;
    }
    // latent
    enc_kernel<<<1, 128, 0, stream>>>(cur, encW, encb, zbuf);
    dec_kernel<<<(unsigned)((16 * 6016 + 255) / 256), 256, 0, stream>>>(zbuf, decW, decb, bufA);

    // ------------------------- decoder -------------------------
    pool(um[3], bufA, bufB, NSh[3], NSh[4], 128);
    conv(bufB, 3, Wd[0], bd[0], 128, 128, bufA, true);
    pool(um[2], bufA, bufB, NSh[2], NSh[3], 128);
    conv(bufB, 2, Wd[1], bd[1], 128, 64, bufA, true);
    pool(um[1], bufA, bufB, NSh[1], NSh[2], 64);
    conv(bufB, 1, Wd[2], bd[2], 64, 32, bufA, true);
    pool(um[0], bufA, bufB, NSh[0], NSh[1], 32);
    conv(bufB, 0, Wd[3], bd[3], 32, 3, (float*)d_out, false);
}
